// ROIGridPooler_52888227283661
// MI455X (gfx1250) — matrix operand order repaired and checked by reading, not compiled, not run on hardware
//
#include <hip/hip_runtime.h>

// ---------------------------------------------------------------------------
// ROI grid pooling (VoteNet-style) for MI455X / gfx1250 (wave32, WMMA)
// ---------------------------------------------------------------------------
typedef __attribute__((ext_vector_type(16))) _Float16 v16h;
typedef __attribute__((ext_vector_type(8)))  _Float16 v8h;
typedef __attribute__((ext_vector_type(8)))  float    v8f;
typedef __attribute__((ext_vector_type(4)))  int      v4i;

#define GRIDD   3
#define KPTS    27           // GRID^3
#define NSAMP   16
#define RAD     0.2f
#define RAD2    (RAD*RAD)
#define NB      2            // B
#define NN      1024         // N seeds
#define NC      256          // C feature channels
#define NP      256          // P proposals
#define NM      (NP*KPTS)    // 6912 keypoints per batch
#define KPAD    288          // layer-1 K padded to 9 chunks of 32

// packed-weight blob (f16 B-fragment blocks, contiguous): w1 | w2 | w3
#define W1_F16  36864u       // 9*8 blocks * 512 f16
#define W2_F16  8192u        // 4*4 blocks * 512 f16
#define W3_F16  2048u        // 2*2 blocks * 512 f16
#define WBLOB_F16 (W1_F16 + W2_F16 + W3_F16)          // 47104 f16 = 94208 B

// workspace byte offsets (all 256B aligned)
#define OFF_KP    0u                                  // B*M*3 f32   = 165888 B
#define OFF_IDX   165888u                             // B*M*16 i32  = 884736 B
#define OFF_WBLOB 1050624u                            // 94208 B
#define OFF_FEATT 1144832u                            // B*N*288 f16 = 1179648 B
                                                      // end 2324480 B

// async global->LDS staging (CDNA5); guarded so absence can't break the build
#if defined(__HIP_DEVICE_COMPILE__) && \
    __has_builtin(__builtin_amdgcn_global_load_async_to_lds_b128) && \
    __has_builtin(__builtin_amdgcn_s_wait_asynccnt)
#define USE_ASYNC_LDS 1
#else
#define USE_ASYNC_LDS 0
#endif

__device__ __forceinline__ v8f vzero8() {
    v8f z = {0.f,0.f,0.f,0.f,0.f,0.f,0.f,0.f};
    return z;
}

// ---------------------------------------------------------------------------
// Kernel 1: rotated grid keypoints. pts = local @ Rz(t) + origin
// ---------------------------------------------------------------------------
__global__ void roi_keypoints_kernel(const float* __restrict__ boxes,
                                     const float* __restrict__ origins,
                                     const float* __restrict__ heading,
                                     float* __restrict__ kpws) {
    int tid = blockIdx.x * blockDim.x + threadIdx.x;
    if (tid >= NB*NM) return;
    int b  = tid / NM;
    int m  = tid % NM;
    int p  = m / KPTS;
    int kk = m % KPTS;
    int gi = kk / 9, gj = (kk / 3) % 3, gk = kk % 3;      // meshgrid 'ij'
    const float* bx = boxes + ((size_t)b*NP + p)*6;
    float sx = bx[0]+bx[3], sy = bx[1]+bx[4], sz = bx[2]+bx[5];
    float lx = (gi + 0.5f) * (1.f/GRIDD) * sx - bx[0];
    float ly = (gj + 0.5f) * (1.f/GRIDD) * sy - bx[1];
    float lz = (gk + 0.5f) * (1.f/GRIDD) * sz - bx[2];
    float t  = heading[b*NP + p];
    float c  = cosf(t), s = sinf(t);
    const float* org = origins + ((size_t)b*NP + p)*3;
    float px =  lx*c + ly*s + org[0];
    float py = -lx*s + ly*c + org[1];
    float pz =  lz          + org[2];
    float* o = kpws + (size_t)tid*3;
    o[0] = px; o[1] = py; o[2] = pz;
}

// ---------------------------------------------------------------------------
// Kernel 2: ball query — first NSAMP seed indices with d2 < R^2, fill rule
// matches top_k(-cand) semantics. Seeds staged in LDS (12 KB).
// ---------------------------------------------------------------------------
__global__ void roi_ballquery_kernel(const float* __restrict__ kpws,
                                     const float* __restrict__ sxyz,
                                     int* __restrict__ idxws) {
    __shared__ float s[NN*3];
    int tid = blockIdx.x * 256 + threadIdx.x;
    int b   = tid / NM;                       // NM % 256 == 0 -> uniform per block
    for (int j = threadIdx.x; j < NN*3; j += 256)
        s[j] = sxyz[(size_t)b*NN*3 + j];
    __syncthreads();

    const float* kp = kpws + (size_t)tid*3;
    float kx = kp[0], ky = kp[1], kz = kp[2];
    int* outp = idxws + (size_t)tid*NSAMP;
    int cnt = 0, first = 0;
    for (int n = 0; n < NN; n++) {
        float dx = kx - s[n*3+0];
        float dy = ky - s[n*3+1];
        float dz = kz - s[n*3+2];
        float d2 = dx*dx + dy*dy + dz*dz;
        if (d2 < RAD2) {
            if (cnt == 0) first = n;
            outp[cnt] = n;
            cnt++;
            if (cnt == NSAMP) break;
        }
    }
    int fill = (cnt == 0) ? 0 : first;
    for (int j = cnt; j < NSAMP; j++) outp[j] = fill;
}

// ---------------------------------------------------------------------------
// Kernel 3: pack fp32 weights (Kin x Ncols, row-major) into f16 WMMA
// B-fragment blocks: block (nt,kc) = 32x16 tile; lane L holds 16 f16
// at k_local = (e>>3)*16 + (L>>4)*8 + (e&7), n = nt*16 + (L&15). Block = 1024 B.
// ---------------------------------------------------------------------------
__global__ void roi_pack_kernel(const float* __restrict__ W, _Float16* __restrict__ Wp,
                                int Kin, int Ncols, int kchunks, int ntiles) {
    int tid   = blockIdx.x * blockDim.x + threadIdx.x;
    int total = kchunks * ntiles * 512;
    if (tid >= total) return;
    int blk    = tid >> 9;
    int within = tid & 511;
    int lane   = within >> 4;
    int e      = within & 15;
    int nt = blk / kchunks;
    int kc = blk % kchunks;
    int kg = kc*32 + ((e >> 3) << 4) + ((lane >> 4) << 3) + (e & 7);
    int n  = nt*16 + (lane & 15);
    float v = (kg < Kin) ? W[(size_t)kg * Ncols + n] : 0.f;
    Wp[(size_t)blk*512 + (size_t)lane*16 + e] = (_Float16)v;
}

// ---------------------------------------------------------------------------
// Kernel 4: transpose seed_features (B,C,N) -> f16 rows (B,N,288) laid out in
// MLP-input order: k=0..2 zero (xyz slot), k=3..258 channels, tail zero.
// Makes the per-sample gather two contiguous b128 loads per K-chunk.
// ---------------------------------------------------------------------------
__global__ void roi_feat_transpose_kernel(const float* __restrict__ sfeat,
                                          _Float16* __restrict__ featT) {
    int tid = blockIdx.x * blockDim.x + threadIdx.x;
    if (tid >= NB*NN*KPAD) return;
    int k  = tid % KPAD;
    int bn = tid / KPAD;
    int n  = bn % NN;
    int b  = bn / NN;
    float v = 0.f;
    int c = k - 3;
    if (c >= 0 && c < NC) v = sfeat[((size_t)b*NC + c)*NN + n];
    featT[tid] = (_Float16)v;
}

// ---------------------------------------------------------------------------
// Kernel 5: fused gather + 3-layer MLP (WMMA f16->f32) + sample-max + GEMV.
// 1 workgroup = 1 proposal, 4 waves; wave w handles keypoints w, w+4, ...
// Each keypoint's 16 samples form one 16-row A tile (M=16 exactly).
// All packed weights staged in LDS once per WG (async global->LDS).
// ---------------------------------------------------------------------------
__global__ void __launch_bounds__(128)
roi_mlp_kernel(const float* __restrict__ sxyz, const _Float16* __restrict__ featT,
               const float* __restrict__ kpws, const int* __restrict__ idxws,
               const _Float16* __restrict__ wblob,
               const float* __restrict__ b1, const float* __restrict__ b2,
               const float* __restrict__ b3,
               const float* __restrict__ Wr, const float* __restrict__ br,
               float* __restrict__ out) {
    __shared__ __align__(16) _Float16 wlds[WBLOB_F16];  // 94208 B packed weights
    __shared__ __align__(16) _Float16 x1s[4][16*128];   // 16 KB, per-wave relu(x1)
    __shared__ __align__(16) _Float16 x2s[4][16*64];    // 8 KB,  per-wave relu(x2)
    __shared__ float fbuf[32*KPTS];                     // 3456 B, pooled feats (transposed)

    const int bid  = blockIdx.x;
    const int b    = bid >> 8;              // NP == 256
    const int p    = bid & 255;
    const int tid  = threadIdx.x;
    const int wid  = tid >> 5;
    const int lane = tid & 31;
    const int nn   = lane & 15;             // N-col / A-row within tile
    const int h    = lane >> 4;             // lane half

    // ---- stage packed-weight blob into LDS (one 94 KB copy per WG) -------
#if USE_ASYNC_LDS
    for (int i = tid; i < (int)(WBLOB_F16/8); i += 128) {
        __builtin_amdgcn_global_load_async_to_lds_b128(
            (__attribute__((address_space(1))) v4i*)(wblob + (size_t)i*8),
            (__attribute__((address_space(3))) v4i*)(wlds + (size_t)i*8),
            0, 0);
    }
    __builtin_amdgcn_s_wait_asynccnt(0);
#else
    for (int i = tid; i < (int)(WBLOB_F16/8); i += 128)
        ((v8h*)wlds)[i] = ((const v8h*)wblob)[i];
#endif
    __syncthreads();

    const _Float16* w1l = wlds;
    const _Float16* w2l = wlds + W1_F16;
    const _Float16* w3l = wlds + W1_F16 + W2_F16;

    for (int kp = wid; kp < KPTS; kp += 4) {          // wave-uniform trip count
        const int m = p * KPTS + kp;
        const float* kpt = kpws + ((size_t)b*NM + m)*3;
        const float kx = kpt[0], ky = kpt[1], kz = kpt[2];
        const int   ns = idxws[((size_t)b*NM + m)*NSAMP + nn];  // sample = nn
        const float* g = sxyz + ((size_t)b*NN + ns)*3;
        const _Float16 axh = (_Float16)((g[0]-kx)*(1.f/RAD));
        const _Float16 ayh = (_Float16)((g[1]-ky)*(1.f/RAD));
        const _Float16 azh = (_Float16)((g[2]-kz)*(1.f/RAD));
        const _Float16* fRow = featT + ((size_t)b*NN + ns) * KPAD;

        // -------- layer 1: (16 x 259) @ (259 x 128), K padded to 288 ------
        v8f acc1[8];
        #pragma unroll
        for (int t = 0; t < 8; t++) acc1[t] = vzero8();

        #pragma unroll
        for (int kc = 0; kc < 9; kc++) {
            // A fragment: two contiguous 8-element runs per lane
            v8h lo = *(const v8h*)(fRow + kc*32 + h*8);
            v8h hi = *(const v8h*)(fRow + kc*32 + 16 + h*8);
            v16h a;
            #pragma unroll
            for (int e = 0; e < 8; e++) { a[e] = lo[e]; a[e+8] = hi[e]; }
            if (kc == 0) {                          // xyz slot (k = 0..2)
                if (h == 0) { a[0] = axh; a[1] = ayh; a[2] = azh; }
            }
            #pragma unroll
            for (int nt = 0; nt < 8; nt++) {
                v16h bf = *(const v16h*)(w1l + (size_t)(nt*9 + kc)*512 + lane*16);
                acc1[nt] = __builtin_amdgcn_wmma_f32_16x16x32_f16(
                    false, a, false, bf, (short)0, acc1[nt], false, false);
            }
        }
        // bias + relu -> LDS (C-layout: vgpr r holds row r + 8*h, col t*16+nn)
        #pragma unroll
        for (int t = 0; t < 8; t++) {
            float bb = b1[t*16 + nn];
            #pragma unroll
            for (int r = 0; r < 8; r++)
                x1s[wid][(r + 8*h)*128 + t*16 + nn] =
                    (_Float16)fmaxf(acc1[t][r] + bb, 0.f);
        }

        // -------- layer 2: (16 x 128) @ (128 x 64) ------------------------
        v8f acc2[4];
        #pragma unroll
        for (int t = 0; t < 4; t++) acc2[t] = vzero8();
        #pragma unroll
        for (int kc = 0; kc < 4; kc++) {
            const _Float16* xr = &x1s[wid][nn*128 + kc*32 + h*8];
            v8h lo = *(const v8h*)(xr);
            v8h hi = *(const v8h*)(xr + 16);
            v16h a;
            #pragma unroll
            for (int e = 0; e < 8; e++) { a[e] = lo[e]; a[e+8] = hi[e]; }
            #pragma unroll
            for (int nt = 0; nt < 4; nt++) {
                v16h bf = *(const v16h*)(w2l + (size_t)(nt*4 + kc)*512 + lane*16);
                acc2[nt] = __builtin_amdgcn_wmma_f32_16x16x32_f16(
                    false, a, false, bf, (short)0, acc2[nt], false, false);
            }
        }
        #pragma unroll
        for (int t = 0; t < 4; t++) {
            float bb = b2[t*16 + nn];
            #pragma unroll
            for (int r = 0; r < 8; r++)
                x2s[wid][(r + 8*h)*64 + t*16 + nn] =
                    (_Float16)fmaxf(acc2[t][r] + bb, 0.f);
        }

        // -------- layer 3: (16 x 64) @ (64 x 32) --------------------------
        v8f acc3[2];
        #pragma unroll
        for (int t = 0; t < 2; t++) acc3[t] = vzero8();
        #pragma unroll
        for (int kc = 0; kc < 2; kc++) {
            const _Float16* xr = &x2s[wid][nn*64 + kc*32 + h*8];
            v8h lo = *(const v8h*)(xr);
            v8h hi = *(const v8h*)(xr + 16);
            v16h a;
            #pragma unroll
            for (int e = 0; e < 8; e++) { a[e] = lo[e]; a[e+8] = hi[e]; }
            #pragma unroll
            for (int nt = 0; nt < 2; nt++) {
                v16h bf = *(const v16h*)(w3l + (size_t)(nt*2 + kc)*512 + lane*16);
                acc3[nt] = __builtin_amdgcn_wmma_f32_16x16x32_f16(
                    false, a, false, bf, (short)0, acc3[nt], false, false);
            }
        }
        // bias + relu + max over the 16 sample-rows; relu => floor at 0
        #pragma unroll
        for (int t = 0; t < 2; t++) {
            float bb = b3[t*16 + nn];
            float vm = 0.f;
            #pragma unroll
            for (int r = 0; r < 8; r++) vm = fmaxf(vm, acc3[t][r] + bb);
            vm = fmaxf(vm, __shfl_xor(vm, 16, 32));   // combine lane halves
            if (lane < 16)
                fbuf[(t*16 + nn)*KPTS + kp] = vm;     // (32, 27) transposed
        }
    }

    __syncthreads();

    // -------- final: out[b, t, p] = br[t] + dot(Wr[t, :864], fbuf) --------
    {
        float acc = br[tid];
        const float* wr = Wr + (size_t)tid * (32*KPTS);
        for (int j = 0; j < 32*KPTS; j++)
            acc += fbuf[j] * wr[j];
        out[((size_t)b*128 + tid)*NP + p] = acc;
    }
}

// ---------------------------------------------------------------------------
// launcher
// ---------------------------------------------------------------------------
extern "C" void kernel_launch(void* const* d_in, const int* in_sizes, int n_in,
                              void* d_out, int out_size, void* d_ws, size_t ws_size,
                              hipStream_t stream) {
    (void)in_sizes; (void)n_in; (void)out_size; (void)ws_size;

    const float* seed_xyz  = (const float*)d_in[0];
    const float* seed_feat = (const float*)d_in[1];
    const float* boxes     = (const float*)d_in[2];
    const float* origins   = (const float*)d_in[3];
    const float* heading   = (const float*)d_in[4];
    const float* W1        = (const float*)d_in[5];
    const float* b1        = (const float*)d_in[6];
    const float* W2        = (const float*)d_in[7];
    const float* b2        = (const float*)d_in[8];
    const float* W3        = (const float*)d_in[9];
    const float* b3        = (const float*)d_in[10];
    const float* Wr        = (const float*)d_in[11];
    const float* br        = (const float*)d_in[12];
    float* out = (float*)d_out;

    char* ws = (char*)d_ws;
    float*     kpws  = (float*)(ws + OFF_KP);
    int*       idxws = (int*)(ws + OFF_IDX);
    _Float16*  wblob = (_Float16*)(ws + OFF_WBLOB);
    _Float16*  w1p   = wblob;
    _Float16*  w2p   = wblob + W1_F16;
    _Float16*  w3p   = wblob + W1_F16 + W2_F16;
    _Float16*  featT = (_Float16*)(ws + OFF_FEATT);

    roi_keypoints_kernel<<<(NB*NM + 255)/256, 256, 0, stream>>>(boxes, origins, heading, kpws);
    roi_ballquery_kernel<<<(NB*NM)/256, 256, 0, stream>>>(kpws, seed_xyz, idxws);
    roi_pack_kernel<<<(9*8*512 + 255)/256, 256, 0, stream>>>(W1, w1p, 259, 128, 9, 8);
    roi_pack_kernel<<<(4*4*512 + 255)/256, 256, 0, stream>>>(W2, w2p, 128,  64, 4, 4);
    roi_pack_kernel<<<(2*2*512 + 255)/256, 256, 0, stream>>>(W3, w3p,  64,  32, 2, 2);
    roi_feat_transpose_kernel<<<(NB*NN*KPAD + 255)/256, 256, 0, stream>>>(seed_feat, featT);

    roi_mlp_kernel<<<NB*NP, 128, 0, stream>>>(seed_xyz, featT, kpws, idxws,
                                              wblob, b1, b2, b3, Wr, br, out);
}
